// TopicRNN_GCN_53970559041618
// MI455X (gfx1250) — matrix operand order repaired and checked by reading, not compiled, not run on hardware
//
#include <hip/hip_runtime.h>
#include <hip/hip_bf16.h>

#define N_NODES 10000
#define EMBED   300
#define XPAD    320       // EMBED padded to multiple of 32
#define HID     256
#define REP     8
#define KP_BIG  10016     // N_NODES padded to multiple of 32 (Bt row stride)

typedef __attribute__((ext_vector_type(16))) __bf16         v16bf;
typedef __attribute__((ext_vector_type(8)))  float          v8f;
typedef __attribute__((ext_vector_type(16))) unsigned short v16u;
typedef __attribute__((ext_vector_type(8)))  unsigned short v8u;
typedef __attribute__((ext_vector_type(4)))  unsigned short v4u;

__device__ __forceinline__ void split_f32(float f, unsigned short& h, unsigned short& l) {
    unsigned u  = __float_as_uint(f);
    unsigned short hs = (unsigned short)(u >> 16);           // truncated bf16 "hi"
    float hf = __uint_as_float((unsigned)hs << 16);
    float r  = f - hf;                                        // exact residual
    l = (unsigned short)(__float_as_uint(r) >> 16);           // bf16 "lo"
    h = hs;
}

// ---------------------------------------------------------------------------
// x[i][e] = max_{r<REP} emb[nodes[(i+1)*REP + r]][e], zero-padded to XPAD cols
// ---------------------------------------------------------------------------
__global__ __launch_bounds__(XPAD) void gather_max_kernel(
    const int* __restrict__ nodes, const float* __restrict__ emb,
    float* __restrict__ x)
{
    __shared__ int idx[REP];
    const int i = blockIdx.x;
    const int e = threadIdx.x;
    if (e < REP) idx[e] = nodes[(size_t)(i + 1) * REP + e];
    __syncthreads();
    float v = 0.0f;
    if (e < EMBED) {
        v = emb[(size_t)idx[0] * EMBED + e];
        #pragma unroll
        for (int r = 1; r < REP; ++r)
            v = fmaxf(v, emb[(size_t)idx[r] * EMBED + e]);
    }
    x[(size_t)i * XPAD + e] = v;
}

// ---------------------------------------------------------------------------
// Bt_hi/Bt_lo[n][k] = split(W[k][n]); k >= K zero-filled (row stride KPp)
// ---------------------------------------------------------------------------
__global__ __launch_bounds__(256) void wsplit_kernel(
    const float* __restrict__ W, int K, int KPp,
    unsigned short* __restrict__ oh, unsigned short* __restrict__ ol)
{
    const int t = blockIdx.x * 256 + threadIdx.x;
    if (t >= HID * KPp) return;
    const int n = t / KPp, k = t % KPp;
    float v = (k < K) ? W[(size_t)k * HID + n] : 0.0f;
    unsigned short h, l;
    split_f32(v, h, l);
    oh[t] = h;
    ol[t] = l;
}

// ---------------------------------------------------------------------------
// C(M x 256) = A(M x Kreal, f32, split on the fly) @ Bt^T (bf16 hi/lo, N x KP)
// bf16x3 product: Ahi*Bhi + Alo*Bhi + Ahi*Blo, f32 accumulate (WMMA 16x16x32).
//
// Workgroup: 8 waves; wave w computes rows [m0,m0+32) x cols [32w,32w+32).
// The shared 32x32 A tile is staged through LDS ONCE per workgroup per K-step,
// split to bf16 hi/lo cooperatively (4 elements/thread), stored pre-swizzled
// in WMMA A-fragment order so each wave reads its fragment as one contiguous
// 32B ds_load. Double-buffered, 1 barrier per K-step. All four B fragments
// are loaded into independent registers up front so the scheduler can clause
// the 8 global_load_b128 and retire them with staged s_wait_loadcnt under the
// full WMMA block (instead of load->wait0->use serialization).
//
// EPI==0: store split(C)^T into Obth/Obtl (N x OKP)  [feeds next GEMM's B]
// EPI==1: C = leaky_relu(C + bias), store f32 row-major (ldc = 256)
// ---------------------------------------------------------------------------
template <int EPI>
__global__ __launch_bounds__(256) void gemm_bf16x3(
    const float* __restrict__ A, int lda, int M, int Kreal, int Kiters,
    const unsigned short* __restrict__ Bth, const unsigned short* __restrict__ Btl, int KP,
    const float* __restrict__ bias, float* __restrict__ Cf,
    unsigned short* __restrict__ Obth, unsigned short* __restrict__ Obtl, int OKP)
{
    // A tile in fragment order: [buf][ (mf*32 + lane)*16 + j ], j = frag element
    __shared__ unsigned short lds_h[2][1024];
    __shared__ unsigned short lds_l[2][1024];

    const int tid  = threadIdx.x;
    const int lane = tid & 31;
    const int wave = tid >> 5;
    const int h    = lane >> 4;   // lane half per ISA 16x32 bf16 A layout
    const int lm   = lane & 15;
    const int m0   = blockIdx.x * 32;
    const int n0   = wave * 32;

    // ---- phase-1 (cooperative A staging) per-thread mapping ----
    // thread t handles tile element (row = t>>3, k = (t&7)*4 .. +4)
    const int p_row = tid >> 3;
    const int p_k   = (tid & 7) * 4;
    const int p_h   = (p_k >> 4) & 1;
    const int p_j   = p_k & 15;
    const int p_mf  = p_row >> 4;
    const int p_ln  = (p_row & 15) + p_h * 16;
    const int p_off = (p_mf * 32 + p_ln) * 16 + p_j;     // v4u-aligned (8B)
    int prow_g = m0 + p_row; if (prow_g > M - 1) prow_g = M - 1;
    const float* p_arow = A + (size_t)prow_g * lda + p_k;

    // ---- B fragment pointers (lane = column, 16 contiguous K, bf16) ----
    const unsigned short* pbh[2];
    const unsigned short* pbl[2];
    pbh[0] = Bth + (size_t)(n0 + lm) * KP + h * 16;
    pbh[1] = Bth + (size_t)(n0 + 16 + lm) * KP + h * 16;
    pbl[0] = Btl + (size_t)(n0 + lm) * KP + h * 16;
    pbl[1] = Btl + (size_t)(n0 + 16 + lm) * KP + h * 16;

    v8f acc[2][2] = {};

    // ---- preload tile 0 into buffer 0 ----
    {
        float4 q{0.f, 0.f, 0.f, 0.f};
        if (p_k < Kreal) q = *(const float4*)(p_arow);
        v4u hq, lq;
        unsigned short hs, ls;
        split_f32(q.x, hs, ls); hq[0] = hs; lq[0] = ls;
        split_f32(q.y, hs, ls); hq[1] = hs; lq[1] = ls;
        split_f32(q.z, hs, ls); hq[2] = hs; lq[2] = ls;
        split_f32(q.w, hs, ls); hq[3] = hs; lq[3] = ls;
        *(v4u*)&lds_h[0][p_off] = hq;
        *(v4u*)&lds_l[0][p_off] = lq;
    }
    __syncthreads();

    for (int it = 0; it < Kiters; ++it) {
        const int  k0  = it * 32;
        const int  cur = it & 1;
        const bool have_next = (it + 1 < Kiters);

        // ---- issue ALL B fragment loads up front (independent registers,
        //      one s_clause of 8 global_load_b128, staged waits) ----
        v16bf bh0 = *(const v16bf*)(pbh[0] + k0);
        v16bf bl0 = *(const v16bf*)(pbl[0] + k0);
        v16bf bh1 = *(const v16bf*)(pbh[1] + k0);
        v16bf bl1 = *(const v16bf*)(pbl[1] + k0);

        // issue next A tile's global loads early (latency hidden under WMMAs)
        float4 qn{0.f, 0.f, 0.f, 0.f};
        if (have_next) {
            const int kn = k0 + 32;
            if (kn + p_k < Kreal) qn = *(const float4*)(p_arow + kn);
            if (kn + 260 <= Kreal) __builtin_prefetch(p_arow + kn + 256, 0, 0);
        }

        // A fragments: one contiguous 32B ds read per frag, direct bit_cast
        v16bf ah[2], al[2];
        #pragma unroll
        for (int mf = 0; mf < 2; ++mf) {
            ah[mf] = __builtin_bit_cast(v16bf, *(const v16u*)&lds_h[cur][(mf * 32 + lane) * 16]);
            al[mf] = __builtin_bit_cast(v16bf, *(const v16u*)&lds_l[cur][(mf * 32 + lane) * 16]);
        }

        // ---- 12 WMMAs: bf16x3 product for 2 m-frags x 2 n-frags ----
        #pragma unroll
        for (int mf = 0; mf < 2; ++mf) {
            acc[mf][0] = __builtin_amdgcn_wmma_f32_16x16x32_bf16(
                false, ah[mf], false, bh0, (short)0, acc[mf][0], false, false);
            acc[mf][0] = __builtin_amdgcn_wmma_f32_16x16x32_bf16(
                false, al[mf], false, bh0, (short)0, acc[mf][0], false, false);
            acc[mf][0] = __builtin_amdgcn_wmma_f32_16x16x32_bf16(
                false, ah[mf], false, bl0, (short)0, acc[mf][0], false, false);
        }
        #pragma unroll
        for (int mf = 0; mf < 2; ++mf) {
            acc[mf][1] = __builtin_amdgcn_wmma_f32_16x16x32_bf16(
                false, ah[mf], false, bh1, (short)0, acc[mf][1], false, false);
            acc[mf][1] = __builtin_amdgcn_wmma_f32_16x16x32_bf16(
                false, al[mf], false, bh1, (short)0, acc[mf][1], false, false);
            acc[mf][1] = __builtin_amdgcn_wmma_f32_16x16x32_bf16(
                false, ah[mf], false, bl1, (short)0, acc[mf][1], false, false);
        }

        // split + stage next tile into the other buffer
        if (have_next) {
            v4u hq, lq;
            unsigned short hs, ls;
            split_f32(qn.x, hs, ls); hq[0] = hs; lq[0] = ls;
            split_f32(qn.y, hs, ls); hq[1] = hs; lq[1] = ls;
            split_f32(qn.z, hs, ls); hq[2] = hs; lq[2] = ls;
            split_f32(qn.w, hs, ls); hq[3] = hs; lq[3] = ls;
            *(v4u*)&lds_h[cur ^ 1][p_off] = hq;
            *(v4u*)&lds_l[cur ^ 1][p_off] = lq;
        }
        __syncthreads();
    }

    // ---- epilogue; C/D layout: VGPR r -> row m0+mf*16+h*8+r, col n0+nf*16+lm ----
    #pragma unroll
    for (int nf = 0; nf < 2; ++nf) {
        const int n = n0 + nf * 16 + lm;
        #pragma unroll
        for (int mf = 0; mf < 2; ++mf) {
            const int mbase = m0 + mf * 16 + h * 8;
            if (EPI == 1) {
                const float bz = bias[n];
                #pragma unroll
                for (int r = 0; r < 8; ++r) {
                    const int m = mbase + r;
                    float v = acc[mf][nf][r] + bz;
                    v = (v > 0.0f) ? v : 0.01f * v;
                    if (m < M) Cf[(size_t)m * HID + n] = v;
                }
            } else {
                v8u hv, lv;
                #pragma unroll
                for (int r = 0; r < 8; ++r) {
                    unsigned short hs, ls;
                    split_f32(acc[mf][nf][r], hs, ls);
                    hv[r] = hs; lv[r] = ls;
                }
                *(v8u*)(Obth + (size_t)n * OKP + mbase) = hv;  // transposed store
                *(v8u*)(Obtl + (size_t)n * OKP + mbase) = lv;
            }
        }
    }
}

// ---------------------------------------------------------------------------
extern "C" void kernel_launch(void* const* d_in, const int* in_sizes, int n_in,
                              void* d_out, int out_size, void* d_ws, size_t ws_size,
                              hipStream_t stream) {
    const int*   nodes = (const int*)d_in[0];
    const float* emb   = (const float*)d_in[1];
    const float* adj   = (const float*)d_in[2];
    const float* W1    = (const float*)d_in[3];
    const float* b1    = (const float*)d_in[4];
    const float* W2    = (const float*)d_in[5];
    const float* b2    = (const float*)d_in[6];
    float* out = (float*)d_out;

    // workspace carve-out (256B aligned slices), ~44 MB total
    char*  ws  = (char*)d_ws;
    size_t off = 0;
    auto carve = [&](size_t bytes) -> char* {
        char* p = ws + off;
        off = (off + bytes + 255) & ~(size_t)255;
        return p;
    };
    float*          x    = (float*)         carve((size_t)N_NODES * XPAD * sizeof(float));
    unsigned short* bt1h = (unsigned short*)carve((size_t)HID * KP_BIG * 2);
    unsigned short* bt1l = (unsigned short*)carve((size_t)HID * KP_BIG * 2);
    float*          h1   = (float*)         carve((size_t)N_NODES * HID * sizeof(float));
    unsigned short* bt2h = (unsigned short*)carve((size_t)HID * KP_BIG * 2);
    unsigned short* bt2l = (unsigned short*)carve((size_t)HID * KP_BIG * 2);
    unsigned short* w1h  = (unsigned short*)carve((size_t)HID * XPAD * 2);
    unsigned short* w1l  = (unsigned short*)carve((size_t)HID * XPAD * 2);
    unsigned short* w2h  = (unsigned short*)carve((size_t)HID * HID * 2);
    unsigned short* w2l  = (unsigned short*)carve((size_t)HID * HID * 2);

    const int MB = (N_NODES + 31) / 32;  // 313 row-blocks

    // x = max-pool gather (zero-padded to XPAD)
    gather_max_kernel<<<N_NODES, XPAD, 0, stream>>>(nodes, emb, x);
    // W1 / W2 -> transposed bf16 hi/lo operands
    wsplit_kernel<<<(HID * XPAD + 255) / 256, 256, 0, stream>>>(W1, EMBED, XPAD, w1h, w1l);
    wsplit_kernel<<<(HID * HID + 255) / 256, 256, 0, stream>>>(W2, HID, HID, w2h, w2l);

    // y1^T = (x @ W1)^T  (bf16 hi/lo, ready as B of the big GEMM)
    gemm_bf16x3<0><<<MB, 256, 0, stream>>>(x, XPAD, N_NODES, XPAD, XPAD / 32,
                                           w1h, w1l, XPAD,
                                           nullptr, nullptr, bt1h, bt1l, KP_BIG);
    // h1 = leaky_relu(adj @ y1 + b1)
    gemm_bf16x3<1><<<MB, 256, 0, stream>>>(adj, N_NODES, N_NODES, N_NODES, MB,
                                           bt1h, bt1l, KP_BIG,
                                           b1, h1, nullptr, nullptr, 0);
    // y2^T = (h1 @ W2)^T
    gemm_bf16x3<0><<<MB, 256, 0, stream>>>(h1, HID, N_NODES, HID, HID / 32,
                                           w2h, w2l, HID,
                                           nullptr, nullptr, bt2h, bt2l, KP_BIG);
    // out = leaky_relu(adj @ y2 + b2)
    gemm_bf16x3<1><<<MB, 256, 0, stream>>>(adj, N_NODES, N_NODES, N_NODES, MB,
                                           bt2h, bt2l, KP_BIG,
                                           b2, out, nullptr, nullptr, 0);
}
